// GRASER_76175539962048
// MI455X (gfx1250) — matrix operand order repaired and checked by reading, not compiled, not run on hardware
//
#include <hip/hip_runtime.h>
#include <hip/hip_bf16.h>
#include <math.h>

#define NITEMS 50000
#define H      100
#define BATCH  1024
#define SEQL   50
#define NNZTOT 819200
#define BL     (BATCH*SEQL)
#define NM1    (NITEMS-1)

// padded K dims (multiples of 32)
#define KP1 128     // K=100
#define KP2 224     // K=200
#define KP3 320     // K=300

typedef __attribute__((ext_vector_type(16))) _Float16 v16h;
typedef __attribute__((ext_vector_type(8)))  _Float16 v8h;
typedef __attribute__((ext_vector_type(8)))  float    v8f;

static inline int cdiv(int a, int b){ return (a + b - 1) / b; }

// ---------------- utility kernels ----------------
__global__ void k_zero_f32(float* p, int n){
  int i = blockIdx.x*blockDim.x + threadIdx.x;
  if (i < n) p[i] = 0.0f;
}
// f32 (rows x cols) -> f16 (rows_pad x cols_pad), zero-padded
__global__ void k_cvt_pad(const float* __restrict__ s, _Float16* __restrict__ d,
                          int rows, int cols, int rows_pad, int cols_pad){
  int i = blockIdx.x*blockDim.x + threadIdx.x;
  int total = rows_pad * cols_pad;
  if (i >= total) return;
  int r = i / cols_pad, c = i - r*cols_pad;
  d[i] = (r < rows && c < cols) ? (_Float16)s[(size_t)r*cols + c] : (_Float16)0.0f;
}
// f32 W (K x N) -> f16 BT (Nrows x Kpad), transposed, zero-padded
__global__ void k_wT(const float* __restrict__ W, _Float16* __restrict__ BT,
                     int K, int N, int Kpad, int Nrows){
  int i = blockIdx.x*blockDim.x + threadIdx.x;
  int total = Nrows * Kpad;
  if (i >= total) return;
  int n = i / Kpad, k = i - n*Kpad;
  BT[i] = (n < N && k < K) ? (_Float16)W[(size_t)k*N + n] : (_Float16)0.0f;
}

// ---------------- scatter SpMM (segment_sum of val * e[col]) ----------------
__global__ void k_spmm_scatter(const int* rows, const int* cols, const float* vals,
                               const float* e, float* acc, int total){
  int i = blockIdx.x*blockDim.x + threadIdx.x;
  if (i >= total) return;
  int nz = i / H, h = i - nz*H;
  atomicAdd(&acc[(size_t)rows[nz]*H + h], vals[nz] * e[(size_t)cols[nz]*H + h]);
}
// pruned: only rows < SEQL reach the final output; e1 is padded f16 (ld = KP1)
__global__ void k_spmm_scatter_rows50(const int* rows, const int* cols, const float* vals,
                                      const _Float16* e16, float* acc, int total){
  int i = blockIdx.x*blockDim.x + threadIdx.x;
  if (i >= total) return;
  int nz = i / H, h = i - nz*H;
  int r = rows[nz];
  if (r >= SEQL) return;
  atomicAdd(&acc[r*H + h], vals[nz] * (float)e16[(size_t)cols[nz]*KP1 + h]);
}

// ---------------- WMMA GEMM: C = act(A @ B + bias) ----------------
// A:  M_pad x KPAD f16 row-major (M_pad mult of 16, pads zeroed)
// BT: N_pad x KPAD f16 row-major = B transposed (N_pad mult of 16, pads zeroed)
// one wave per 16x16 tile; K loop fully unrolled; branch-free vector loads;
// fully specialized (branch-free) epilogue.
// ACT: 0=none 1=leaky_relu(0.2) 2=sigmoid 3=tanh
// FULLM: M is a multiple of 16 -> no row guard in epilogue (straight-line stores)
template<int KPAD, int ACT, bool BIAS, bool WF, bool WH, bool FULLM>
__global__ void k_gemm_t(const _Float16* __restrict__ A, const _Float16* __restrict__ BT,
                         const float* __restrict__ bias,
                         float* __restrict__ Cf, _Float16* __restrict__ Ch,
                         int M, int N, int lda, int ldbt, int ldc)
{
  const int lane = threadIdx.x;            // 0..31 (wave32)
  const int half = lane >> 4;              // lane group 0/1
  const int l15  = lane & 15;
  const int m0 = blockIdx.x * 16;
  const int n0 = blockIdx.y * 16;
  const _Float16* arow = A  + (size_t)(m0 + l15) * lda;
  const _Float16* brow = BT + (size_t)(n0 + l15) * ldbt;
  v8f acc = {};

  #pragma unroll
  for (int k0 = 0; k0 < KPAD; k0 += 32) {
    // A 16x32 f16: group0 -> K k0..k0+15, group1 -> k0+16..k0+31
    v16h a = *(const v16h*)(arow + k0 + half*16);
    // B 32x16 f16: lane col = l15, group0 -> K k0..k0+15, group1 -> k0+16..k0+31
    v16h b = *(const v16h*)(brow + k0 + half*16);
    acc = __builtin_amdgcn_wmma_f32_16x16x32_f16(false, a, false, b,
                                                 (short)0, acc, false, false);
  }

  // C/D layout: vgpr r holds row m0 + half*8 + r, column n0 + l15
  int col = n0 + l15;
  if (col < N) {
    float bv = 0.0f;
    if constexpr (BIAS) bv = bias[col];
    #pragma unroll
    for (int r = 0; r < 8; ++r) {
      int row = m0 + half*8 + r;
      if (FULLM || row < M) {
        float v = acc[r];
        if constexpr (BIAS)     v += bv;
        if constexpr (ACT == 1) v = (v > 0.0f) ? v : 0.2f * v;
        if constexpr (ACT == 2) v = 1.0f / (1.0f + __expf(-v));
        if constexpr (ACT == 3) v = tanhf(v);
        if constexpr (WF) Cf[(size_t)row * ldc + col] = v;
        if constexpr (WH) Ch[(size_t)row * ldc + col] = (_Float16)v;
      }
    }
  }
}

// ---------------- small fused kernels ----------------
// cat50: 64 x KP3, [e0 | e1 | e2] rows<50, zero padded
__global__ void k_build_cat50(const _Float16* emb16p, const _Float16* e1_16,
                              const _Float16* e2_16, _Float16* cat){
  int i = blockIdx.x*blockDim.x + threadIdx.x;
  if (i >= 64*KP3) return;
  int row = i / KP3, c = i - row*KP3;
  _Float16 v = (_Float16)0.0f;
  if (row < SEQL) {
    if      (c <   H)  v = emb16p[(size_t)row*KP1 + c];
    else if (c < 2*H)  v = e1_16 [(size_t)row*KP1 + (c - H)];
    else if (c < 3*H)  v = e2_16 [(size_t)row*KP1 + (c - 2*H)];
  }
  cat[i] = v;
}

// xx: BL x KP2, [x | x] zero padded
__global__ void k_build_xx(const float* embedding, const int* item, _Float16* xx, int total){
  int i = blockIdx.x*blockDim.x + threadIdx.x;
  if (i >= total) return;
  int row = i / KP2, c = i - row*KP2;
  _Float16 v = (_Float16)0.0f;
  if (c < 2*H) {
    int h = (c < H) ? c : (c - H);
    v = (_Float16)embedding[(size_t)item[row]*H + h];
  }
  xx[i] = v;
}

__global__ void k_post_gate(const float* embedding, const int* item, const float* g,
                            _Float16* xx, int total){
  int i = blockIdx.x*blockDim.x + threadIdx.x;
  if (i >= total) return;
  int row = i / H, h = i - row*H;
  float r = g[(size_t)row*2*H + h];                    // r = g[:, :H]
  float x = embedding[(size_t)item[row]*H + h];
  xx[(size_t)row*KP2 + H + h] = (_Float16)(r * x);     // second half := r*h (h==x)
}

__global__ void k_re_emb(const float* embedding, const int* item, const float* g,
                         const float* c, float* re, int total){
  int i = blockIdx.x*blockDim.x + threadIdx.x;
  if (i >= total) return;
  int row = i / H, h = i - row*H;
  float u = g[(size_t)row*2*H + H + h];                // u = g[:, H:]
  float x = embedding[(size_t)item[row]*H + h];
  re[i] = u*x + (1.0f - u)*c[i];
}

__global__ void k_last(const float* mask, const int* alias, const float* re, float* last_h){
  int b = blockIdx.x*blockDim.x + threadIdx.x;
  if (b >= BATCH) return;
  int rm = 0;
  for (int l = 0; l < SEQL; ++l) rm += (mask[b*SEQL + l] > 0.5f) ? 1 : 0;
  int lid = alias[b*SEQL + (rm - 1)];
  for (int h = 0; h < H; ++h)
    last_h[b*H + h] = re[((size_t)b*SEQL + lid)*H + h];
}

// seqh: BL x KP1, zero padded
__global__ void k_seqh(const int* alias, const float* re, _Float16* sh, int total){
  int i = blockIdx.x*blockDim.x + threadIdx.x;
  if (i >= total) return;
  int bl = i / KP1, h = i - bl*KP1;
  _Float16 v = (_Float16)0.0f;
  if (h < H) {
    int b = bl / SEQL, l = bl - b*SEQL;
    int a = alias[b*SEQL + l];
    v = (_Float16)re[((size_t)b*SEQL + a)*H + h];
  }
  sh[i] = v;
}

__global__ void k_coef(const float* last_h, const float* seq, const float* nasr_b,
                       const float* nasr_v, const float* mask, float* coef){
  int bl = blockIdx.x*blockDim.x + threadIdx.x;
  if (bl >= BL) return;
  int b = bl / SEQL, l = bl - b*SEQL;
  float s = 0.0f;
  for (int h = 0; h < H; ++h) {
    float m = last_h[b*H + h] + seq[(size_t)bl*H + h] + nasr_b[h];
    m = 1.0f / (1.0f + __expf(-m));
    s += m * nasr_v[h];
  }
  coef[bl] = s * mask[b*SEQL + l];
}

// glb (B x H f32) and last_emb (B x KP1 f16, padded)
__global__ void k_glb_lastemb(const float* coef, const int* alias, const float* re,
                              const int* item, const float* embed50,
                              float* glb, _Float16* lastemb16, int total){
  int i = blockIdx.x*blockDim.x + threadIdx.x;
  if (i >= total) return;               // total = BATCH*KP1
  int b = i / KP1, h = i - b*KP1;
  if (h >= H) { lastemb16[i] = (_Float16)0.0f; return; }
  float gsum = 0.0f, ws = 0.0f, le = 0.0f;
  for (int l = 0; l < SEQL; ++l) {
    int a = alias[b*SEQL + l];
    gsum += coef[b*SEQL + l] * re[((size_t)b*SEQL + a)*H + h];
    float w = (item[b*SEQL + l] != 0) ? 1.0f : 0.0f;
    ws += w;
    le += w * embed50[l*H + h];
  }
  glb[b*H + h] = gsum;
  lastemb16[i] = (_Float16)(le / fmaxf(ws, 1.0f));
}

// ma: B x KP1 f16 padded
__global__ void k_ma(const float* glb, const float* we3, const float* last_h,
                     _Float16* ma16, int total){
  int i = blockIdx.x*blockDim.x + threadIdx.x;
  if (i >= total) return;               // total = BATCH*KP1
  int b = i / KP1, h = i - b*KP1;
  ma16[i] = (h < H) ? (_Float16)((glb[b*H + h] + we3[b*H + h]) * last_h[b*H + h])
                    : (_Float16)0.0f;
}

// ---------------- log-softmax + NLL loss over 49999-wide rows ----------------
__global__ void k_loss(const float* logits, const int* tar, float* loss_out){
  __shared__ float red[256];
  int b = blockIdx.x, t = threadIdx.x;
  const float* row = logits + (size_t)b * NM1;
  float mx = -3.0e38f;
  for (int i = t; i < NM1; i += 256) mx = fmaxf(mx, row[i]);
  red[t] = mx; __syncthreads();
  for (int s = 128; s > 0; s >>= 1){ if (t < s) red[t] = fmaxf(red[t], red[t+s]); __syncthreads(); }
  mx = red[0]; __syncthreads();
  float sum = 0.0f;
  for (int i = t; i < NM1; i += 256) sum += __expf(row[i] - mx);
  red[t] = sum; __syncthreads();
  for (int s = 128; s > 0; s >>= 1){ if (t < s) red[t] += red[t+s]; __syncthreads(); }
  if (t == 0) {
    float lse = mx + __logf(red[0]);
    float lp = row[tar[b] - 1] - lse;
    atomicAdd(loss_out, -lp / (float)BATCH);
  }
}

// ---------------- host ----------------
extern "C" void kernel_launch(void* const* d_in, const int* in_sizes, int n_in,
                              void* d_out, int out_size, void* d_ws, size_t ws_size,
                              hipStream_t stream)
{
  (void)in_sizes; (void)n_in; (void)out_size; (void)ws_size;
  const float* embedding = (const float*)d_in[0];
  const float* adj_vals  = (const float*)d_in[1];
  const float* W_gc0     = (const float*)d_in[2];
  const float* b_gc0     = (const float*)d_in[3];
  const float* W_gc1     = (const float*)d_in[4];
  const float* b_gc1     = (const float*)d_in[5];
  const float* W_concat  = (const float*)d_in[6];
  const float* b_concat  = (const float*)d_in[7];
  const float* gate_k    = (const float*)d_in[8];
  const float* gate_b    = (const float*)d_in[9];
  const float* cand_k    = (const float*)d_in[10];
  const float* cand_b    = (const float*)d_in[11];
  const float* nasr_w2   = (const float*)d_in[12];
  const float* nasr_w3   = (const float*)d_in[13];
  const float* nasr_v    = (const float*)d_in[14];
  const float* nasr_b    = (const float*)d_in[15];
  const float* mask      = (const float*)d_in[16];
  const int*   alias     = (const int*)d_in[17];
  const int*   item      = (const int*)d_in[18];
  const int*   tar       = (const int*)d_in[19];
  const int*   adj_rows  = (const int*)d_in[20];
  const int*   adj_cols  = (const int*)d_in[21];

  char* base = (char*)d_ws;
  size_t off = 0;
  auto take = [&](size_t bytes) -> char* {
    off = (off + 255) & ~(size_t)255;
    char* p = base + off; off += bytes; return p;
  };

  // padded f16 operands
  _Float16* emb16p   = (_Float16*)take((size_t)(NITEMS+16)*KP1*2); // embedding, padded
  float*    acc      = (float*)   take((size_t)NITEMS*H*4);        // spmm(e0), f32
  _Float16* acc16p   = (_Float16*)take((size_t)NITEMS*KP1*2);
  _Float16* e1_16    = (_Float16*)take((size_t)NITEMS*KP1*2);
  _Float16* wgc0T    = (_Float16*)take((size_t)112*KP1*2);
  _Float16* wgc1T    = (_Float16*)take((size_t)112*KP1*2);
  _Float16* wcatT    = (_Float16*)take((size_t)112*KP3*2);
  _Float16* gatekT   = (_Float16*)take((size_t)208*KP2*2);
  _Float16* candkT   = (_Float16*)take((size_t)112*KP2*2);
  _Float16* w2T      = (_Float16*)take((size_t)112*KP1*2);
  _Float16* w3T      = (_Float16*)take((size_t)112*KP1*2);
  float*    acc50    = (float*)   take((size_t)SEQL*H*4);
  _Float16* acc50p   = (_Float16*)take((size_t)64*KP1*2);
  _Float16* e2_16    = (_Float16*)take((size_t)64*KP1*2);
  _Float16* cat50_16 = (_Float16*)take((size_t)64*KP3*2);
  float*    embed50  = (float*)   take((size_t)SEQL*H*4);
  _Float16* xx16     = (_Float16*)take((size_t)BL*KP2*2);          // [x|x] then [x|r*x]
  float*    gbuf     = (float*)   take((size_t)BL*2*H*4);
  float*    cbuf     = (float*)   take((size_t)BL*H*4);
  float*    rebuf    = (float*)   take((size_t)BL*H*4);
  _Float16* seqh16   = (_Float16*)take((size_t)BL*KP1*2);
  float*    seqbuf   = (float*)   take((size_t)BL*H*4);
  float*    coef     = (float*)   take((size_t)BL*4);
  float*    last_h   = (float*)   take((size_t)BATCH*H*4);
  _Float16* lastemb16= (_Float16*)take((size_t)BATCH*KP1*2);
  float*    we3      = (float*)   take((size_t)BATCH*H*4);
  float*    glb      = (float*)   take((size_t)BATCH*H*4);
  _Float16* ma16     = (_Float16*)take((size_t)BATCH*KP1*2);

  float* out_loss   = (float*)d_out;
  float* out_logits = (float*)d_out + 1;

  const int T = 256;
  // --- padded/transposed f16 operand prep ---
  k_cvt_pad<<<cdiv((NITEMS+16)*KP1, T), T, 0, stream>>>(embedding, emb16p,
                                                        NITEMS, H, NITEMS+16, KP1);
  k_wT<<<cdiv(112*KP1, T), T, 0, stream>>>(W_gc0,    wgc0T,  H,   H,   KP1, 112);
  k_wT<<<cdiv(112*KP1, T), T, 0, stream>>>(W_gc1,    wgc1T,  H,   H,   KP1, 112);
  k_wT<<<cdiv(112*KP3, T), T, 0, stream>>>(W_concat, wcatT,  3*H, H,   KP3, 112);
  k_wT<<<cdiv(208*KP2, T), T, 0, stream>>>(gate_k,   gatekT, 2*H, 2*H, KP2, 208);
  k_wT<<<cdiv(112*KP2, T), T, 0, stream>>>(cand_k,   candkT, 2*H, H,   KP2, 112);
  k_wT<<<cdiv(112*KP1, T), T, 0, stream>>>(nasr_w2,  w2T,    H,   H,   KP1, 112);
  k_wT<<<cdiv(112*KP1, T), T, 0, stream>>>(nasr_w3,  w3T,    H,   H,   KP1, 112);

  // --- conv layer 1 (full N) ---
  k_zero_f32<<<cdiv(NITEMS*H, T), T, 0, stream>>>(acc, NITEMS*H);
  k_spmm_scatter<<<cdiv(NNZTOT*H, T), T, 0, stream>>>(adj_rows, adj_cols, adj_vals,
                                                      embedding, acc, NNZTOT*H);
  k_cvt_pad<<<cdiv(NITEMS*KP1, T), T, 0, stream>>>(acc, acc16p, NITEMS, H, NITEMS, KP1);
  { dim3 g(cdiv(NITEMS,16), cdiv(H,16));
    k_gemm_t<KP1,1,true,false,true,true><<<g, 32, 0, stream>>>(acc16p, wgc0T, b_gc0,
        (float*)nullptr, e1_16, NITEMS, H, KP1, KP1, KP1); }

  // --- conv layer 2: only rows < 50 reach the output ---
  k_zero_f32<<<cdiv(SEQL*H, T), T, 0, stream>>>(acc50, SEQL*H);
  k_spmm_scatter_rows50<<<cdiv(NNZTOT*H, T), T, 0, stream>>>(adj_rows, adj_cols, adj_vals,
                                                             e1_16, acc50, NNZTOT*H);
  k_cvt_pad<<<cdiv(64*KP1, T), T, 0, stream>>>(acc50, acc50p, SEQL, H, 64, KP1);
  { dim3 g(cdiv(SEQL,16), cdiv(H,16));
    k_gemm_t<KP1,1,true,false,true,false><<<g, 32, 0, stream>>>(acc50p, wgc1T, b_gc1,
        (float*)nullptr, e2_16, SEQL, H, KP1, KP1, KP1); }

  // --- embed[:50] = lrelu(concat(e0,e1,e2)[:50] @ W_concat + b) ---
  k_build_cat50<<<cdiv(64*KP3, T), T, 0, stream>>>(emb16p, e1_16, e2_16, cat50_16);
  { dim3 g(cdiv(SEQL,16), cdiv(H,16));
    k_gemm_t<KP3,1,true,true,false,false><<<g, 32, 0, stream>>>(cat50_16, wcatT, b_concat,
        embed50, (_Float16*)nullptr, SEQL, H, KP3, KP3, H); }

  // --- GRU step ---
  k_build_xx<<<cdiv(BL*KP2, T), T, 0, stream>>>(embedding, item, xx16, BL*KP2);
  { dim3 g(cdiv(BL,16), cdiv(2*H,16));
    k_gemm_t<KP2,2,true,true,false,true><<<g, 32, 0, stream>>>(xx16, gatekT, gate_b,
        gbuf, (_Float16*)nullptr, BL, 2*H, KP2, KP2, 2*H); }
  k_post_gate<<<cdiv(BL*H, T), T, 0, stream>>>(embedding, item, gbuf, xx16, BL*H);
  { dim3 g(cdiv(BL,16), cdiv(H,16));
    k_gemm_t<KP2,3,true,true,false,true><<<g, 32, 0, stream>>>(xx16, candkT, cand_b,
        cbuf, (_Float16*)nullptr, BL, H, KP2, KP2, H); }
  k_re_emb<<<cdiv(BL*H, T), T, 0, stream>>>(embedding, item, gbuf, cbuf, rebuf, BL*H);

  // --- readout / attention ---
  k_last<<<cdiv(BATCH, T), T, 0, stream>>>(mask, alias, rebuf, last_h);
  k_seqh<<<cdiv(BL*KP1, T), T, 0, stream>>>(alias, rebuf, seqh16, BL*KP1);
  { dim3 g(cdiv(BL,16), cdiv(H,16));
    k_gemm_t<KP1,0,false,true,false,true><<<g, 32, 0, stream>>>(seqh16, w2T, (const float*)nullptr,
        seqbuf, (_Float16*)nullptr, BL, H, KP1, KP1, H); }
  k_coef<<<cdiv(BL, T), T, 0, stream>>>(last_h, seqbuf, nasr_b, nasr_v, mask, coef);
  k_glb_lastemb<<<cdiv(BATCH*KP1, T), T, 0, stream>>>(coef, alias, rebuf, item, embed50,
                                                      glb, lastemb16, BATCH*KP1);
  { dim3 g(cdiv(BATCH,16), cdiv(H,16));
    k_gemm_t<KP1,0,false,true,false,true><<<g, 32, 0, stream>>>(lastemb16, w3T, (const float*)nullptr,
        we3, (_Float16*)nullptr, BATCH, H, KP1, KP1, H); }
  k_ma<<<cdiv(BATCH*KP1, T), T, 0, stream>>>(glb, we3, last_h, ma16, BATCH*KP1);

  // --- logits = ma @ embedding[1:].T ; BT = emb16p rows 1..50000 (contiguous K reads) ---
  { dim3 g(cdiv(BATCH,16), cdiv(NM1,16));
    k_gemm_t<KP1,0,false,true,false,true><<<g, 32, 0, stream>>>(ma16, emb16p + KP1,
        (const float*)nullptr, out_logits, (_Float16*)nullptr, BATCH, NM1, KP1, KP1, NM1); }

  // --- loss = -mean(log_softmax(logits)[b, tar-1]) ---
  k_zero_f32<<<1, 32, 0, stream>>>(out_loss, 1);
  k_loss<<<BATCH, 256, 0, stream>>>(out_logits, tar, out_loss);
}
